// CachedAttention_12824772346386
// MI455X (gfx1250) — compile-verified, matrix operand order checked by reading, not run on hardware
//
#include <hip/hip_runtime.h>

// ---------------------------------------------------------------------------
// CDNA5 (gfx1250) cached-attention prefill, f16 WMMA everywhere.
// b=2, s=2048, dim=4096, H=32, KVH=8, hd=128, max_len=4096, start_pos=0.
// ---------------------------------------------------------------------------

typedef _Float16 v16h __attribute__((ext_vector_type(16)));
typedef _Float16 v8h  __attribute__((ext_vector_type(8)));
typedef float    vf8  __attribute__((ext_vector_type(8)));

#define WMMA_F16(a, b, c) \
    __builtin_amdgcn_wmma_f32_16x16x32_f16(false, (a), false, (b), (short)0, (c), false, false)

// Load one 16-half A/B fragment chunk pair per the CDNA5 16-bit A layout:
// lanes 0-15 hold K = [0..7, 16..23], lanes 16-31 hold K = [8..15, 24..31].
// Caller passes p = row_base + kg  where kg = (lane>>4)*8; reads p and p+16.
__device__ __forceinline__ v16h ldfrag(const _Float16* __restrict__ p) {
    v8h lo = *(const v8h*)(p);
    v8h hi = *(const v8h*)(p + 16);
    return __builtin_shufflevector(lo, hi, 0,1,2,3,4,5,6,7,8,9,10,11,12,13,14,15);
}

// Per-lane async 16B copy global -> LDS (ASYNCcnt-tracked, no VGPR round-trip).
__device__ __forceinline__ void async_cp16(unsigned lds_off, const _Float16* g) {
    asm volatile("global_load_async_to_lds_b128 %0, %1, off"
                 :: "v"(lds_off), "v"((unsigned long long)(size_t)g)
                 : "memory");
}

// ---------------------------------------------------------------------------
// f32 -> f16 conversion (vectorized x4)
// ---------------------------------------------------------------------------
__global__ void __launch_bounds__(256)
cvt_f32_f16(const float* __restrict__ s, _Float16* __restrict__ d, int n4) {
    int i = blockIdx.x * blockDim.x + threadIdx.x;
    if (i >= n4) return;
    const float4 v = ((const float4*)s)[i];
    _Float16* dp = d + i * 4;
    dp[0] = (_Float16)v.x; dp[1] = (_Float16)v.y;
    dp[2] = (_Float16)v.z; dp[3] = (_Float16)v.w;
}

// ---------------------------------------------------------------------------
// Epilogue index helpers (hard-coded to this problem's shapes).
// f32 mode 1 : KV-cache layout  [b,8,4096,128] from (row=(b,s), col=(kvh,d))
// f16 mode 2 : V-transposed     [b,8,128,2048]
// ---------------------------------------------------------------------------
__device__ __forceinline__ int cf_index(int mode, int m, int n, int N) {
    if (mode == 0) return m * N + n;
    int b = m >> 11, s = m & 2047, kvh = n >> 7, d = n & 127;
    return (((b << 3) + kvh) * 4096 + s) * 128 + d;
}
__device__ __forceinline__ int ch_index(int mode, int m, int n, int N) {
    if (mode == 0) return m * N + n;
    int b = m >> 11, s = m & 2047, kvh = n >> 7, d = n & 127;
    return ((((b << 3) + kvh) * 128) + d) * 2048 + s;
}

// ---------------------------------------------------------------------------
// C[M,N] = X[M,K] * W[N,K]^T, f16 in, fp32 accumulate.
// Block = 8 waves (2M x 4N) -> 64x128 C tile; wave = 32x32 (2x2 WMMA tiles).
// A (X) tile double-buffered in LDS via per-lane global_load_async_to_lds_b128;
// B (W) fragments from global (L2-resident) with prefetch.
// LDS rows padded to 40 halves (80B) for conflict-free ds_load_b128 frags.
// ---------------------------------------------------------------------------
#define AROW 40
__global__ void __launch_bounds__(256)
gemm_xwt(const _Float16* __restrict__ X, const _Float16* __restrict__ W,
         float* __restrict__ Cf, int cf_mode,
         _Float16* __restrict__ Ch, int ch_mode,
         int M, int N, int K)
{
    __shared__ _Float16 abuf[2][64 * AROW];   // 2 x 5.0 KB

    const int lane = threadIdx.x & 31;
    const int w    = threadIdx.x >> 5;
    const int bn   = blockIdx.x % (N >> 7);
    const int bm   = blockIdx.x / (N >> 7);
    const int tmB  = bm << 6;                 // block A rows
    const int tn   = (bn << 7) + ((w & 3) << 5);
    const int am   = (w >> 2) << 5;           // wave A-row offset in block tile
    const int r    = lane & 15;
    const int kg   = (lane >> 4) << 3;        // 0 or 8 halves

    // Cooperative A staging: 256 threads x 16B = 64 rows x 64B.
    const int crow = threadIdx.x >> 2;
    const int cch  = threadIdx.x & 3;
    const _Float16* gA = X + (size_t)(tmB + crow) * K + cch * 8;
    const unsigned lA0 = (unsigned)(size_t)&abuf[0][crow * AROW + cch * 8];
    const unsigned lA1 = (unsigned)(size_t)&abuf[1][crow * AROW + cch * 8];

    const _Float16* wb0 = W + (size_t)(tn + r)      * K + kg;
    const _Float16* wb1 = W + (size_t)(tn + 16 + r) * K + kg;

    vf8 c00 = {}, c01 = {}, c10 = {}, c11 = {};

    async_cp16(lA0, gA);                      // stage k0 = 0 into buffer 0

    for (int k0 = 0, i = 0; k0 < K; k0 += 32, ++i) {
        const int cur = i & 1;
        const bool more = (k0 + 32) < K;
        if (more) async_cp16(cur ? lA0 : lA1, gA + k0 + 32);   // prefetch next

        v16h b0 = ldfrag(wb0 + k0);
        v16h b1 = ldfrag(wb1 + k0);
        if (k0 + 96 < K) {
            __builtin_prefetch(wb0 + k0 + 96, 0, 0);
            __builtin_prefetch(wb1 + k0 + 96, 0, 0);
        }

        if (more) asm volatile("s_wait_asynccnt 0x1" ::: "memory");
        else      asm volatile("s_wait_asynccnt 0x0" ::: "memory");
        __syncthreads();                      // all waves' A-tile parts landed

        const _Float16* ab = abuf[cur];
        v16h a0 = ldfrag(ab + (am + r)      * AROW + kg);
        v16h a1 = ldfrag(ab + (am + 16 + r) * AROW + kg);

        c00 = WMMA_F16(a0, b0, c00);
        c01 = WMMA_F16(a0, b1, c01);
        c10 = WMMA_F16(a1, b0, c10);
        c11 = WMMA_F16(a1, b1, c11);

        __syncthreads();                      // reads done before next overwrite
    }

    // C layout: element (m,n): n = lane&15 (+0/16 col tile), m = vgpr + 8*(lane>>4)
    const int mofs = (lane >> 4) << 3;
    const int tm   = tmB + am;
#pragma unroll
    for (int j = 0; j < 8; ++j) {
        const int m   = j + mofs;
        const int gm0 = tm + m, gm1 = tm + 16 + m;
        const int gn0 = tn + r, gn1 = tn + 16 + r;
        if (Cf) {
            Cf[cf_index(cf_mode, gm0, gn0, N)] = c00[j];
            Cf[cf_index(cf_mode, gm0, gn1, N)] = c01[j];
            Cf[cf_index(cf_mode, gm1, gn0, N)] = c10[j];
            Cf[cf_index(cf_mode, gm1, gn1, N)] = c11[j];
        }
        if (Ch) {
            Ch[ch_index(ch_mode, gm0, gn0, N)] = (_Float16)c00[j];
            Ch[ch_index(ch_mode, gm0, gn1, N)] = (_Float16)c01[j];
            Ch[ch_index(ch_mode, gm1, gn0, N)] = (_Float16)c10[j];
            Ch[ch_index(ch_mode, gm1, gn1, N)] = (_Float16)c11[j];
        }
    }
}

// ---------------------------------------------------------------------------
// Zero cache positions [2048, 4096) for both caches (d_out is poisoned).
// ---------------------------------------------------------------------------
__global__ void __launch_bounds__(256)
zero_tail(float* __restrict__ ck, float* __restrict__ cv) {
    int i = blockIdx.x * blockDim.x + threadIdx.x;     // 2 * 16 * 2048 * 128
    if (i >= 8388608) return;
    float* base = (i < 4194304) ? ck : cv;
    int j   = i & 4194303;
    int d   = j & 127;
    int pos = 2048 + ((j >> 7) & 2047);
    int bk  = j >> 18;                                  // (b*8 + kvh)
    base[((size_t)bk * 4096 + pos) * 128 + d] = 0.0f;
}

// ---------------------------------------------------------------------------
// Flash-style attention: one wave = 16 queries of one (b, h).
// Qh : [b*2048, 4096] f16 (row-major, head offset h*128)
// Kh : [b*2048, 1024] f16 (row-major, kv-head offset kvh*128)
// Vt : [b, 8, 128, 2048] f16 (transposed V)
// Att: [b*2048, 4096] f16 output (pre out-projection)
// ---------------------------------------------------------------------------
__global__ void __launch_bounds__(256)
attn_fa(const _Float16* __restrict__ Qh, const _Float16* __restrict__ Kh,
        const _Float16* __restrict__ Vt, _Float16* __restrict__ Att)
{
    __shared__ _Float16 lds[8][16][32];   // per-wave P staging (16q x 32k)

    const int lane = threadIdx.x & 31;
    const int w    = threadIdx.x >> 5;
    const int wid  = blockIdx.x * 8 + w;
    const int qt   = wid & 127;
    const int h    = (wid >> 7) & 31;
    const int b    = wid >> 12;
    const int kvh  = h >> 2;
    const int r    = lane & 15;
    const int kg   = (lane >> 4) << 3;    // doubles as row offset m = j + kg in C layout

    const _Float16* Qb  = Qh + ((size_t)(b * 2048 + qt * 16 + r)) * 4096 + h * 128;
    const _Float16* Kb  = Kh + ((size_t)(b * 2048)) * 1024 + kvh * 128;
    const _Float16* Vtb = Vt + ((size_t)(b * 8 + kvh) * 128) * 2048;

    v16h qa[4];
#pragma unroll
    for (int t = 0; t < 4; ++t) qa[t] = ldfrag(Qb + 32 * t + kg);

    float mi[8], li[8];
    vf8 o[8] = {};
#pragma unroll
    for (int j = 0; j < 8; ++j) { mi[j] = -1e30f; li[j] = 0.0f; }

    const float sc = 0.08838834764831845f;   // 1/sqrt(128)
    const int kmax = qt * 16 + 16;           // exclusive causal bound

    for (int s0 = 0; s0 < kmax; s0 += 32) {
        vf8 S0 = {}, S1 = {};
#pragma unroll
        for (int t = 0; t < 4; ++t) {
            v16h bk0 = ldfrag(Kb + (size_t)(s0 + r)      * 1024 + 32 * t + kg);
            v16h bk1 = ldfrag(Kb + (size_t)(s0 + 16 + r) * 1024 + 32 * t + kg);
            S0 = WMMA_F16(qa[t], bk0, S0);
            S1 = WMMA_F16(qa[t], bk1, S1);
        }

        // Online softmax, all in C-register layout: row m = j + kg, col = r.
#pragma unroll
        for (int j = 0; j < 8; ++j) {
            const int m     = j + kg;
            const int q_idx = qt * 16 + m;
            float s0j = S0[j] * sc; if (s0 + r      > q_idx) s0j = -1e30f;
            float s1j = S1[j] * sc; if (s0 + 16 + r > q_idx) s1j = -1e30f;

            float rm = fmaxf(s0j, s1j);
            rm = fmaxf(rm, __shfl_xor(rm, 1, 16));
            rm = fmaxf(rm, __shfl_xor(rm, 2, 16));
            rm = fmaxf(rm, __shfl_xor(rm, 4, 16));
            rm = fmaxf(rm, __shfl_xor(rm, 8, 16));

            const float nm = fmaxf(mi[j], rm);
            const float ef = __expf(mi[j] - nm);
            const float p0 = __expf(s0j - nm);
            const float p1 = __expf(s1j - nm);

            float rs = p0 + p1;
            rs += __shfl_xor(rs, 1, 16);
            rs += __shfl_xor(rs, 2, 16);
            rs += __shfl_xor(rs, 4, 16);
            rs += __shfl_xor(rs, 8, 16);

            li[j] = li[j] * ef + rs;
            mi[j] = nm;
#pragma unroll
            for (int t = 0; t < 8; ++t) o[t][j] *= ef;

            lds[w][m][r]      = (_Float16)p0;
            lds[w][m][16 + r] = (_Float16)p1;
        }

        // Per-wave LDS round-trip: C-layout -> A-fragment reshape of P.
        asm volatile("s_wait_dscnt 0" ::: "memory");
        v16h pa = ldfrag(&lds[w][r][kg]);

        // O += P * V  via V^T rows (contiguous key chunks).
#pragma unroll
        for (int t = 0; t < 8; ++t) {
            v16h bv = ldfrag(Vtb + (size_t)(t * 16 + r) * 2048 + s0 + kg);
            o[t] = WMMA_F16(pa, bv, o[t]);
        }
    }

    _Float16* A = Att + ((size_t)(b * 2048 + qt * 16)) * 4096 + h * 128;
#pragma unroll
    for (int j = 0; j < 8; ++j) {
        const int m = j + kg;
        const float inv = 1.0f / li[j];
#pragma unroll
        for (int t = 0; t < 8; ++t)
            A[(size_t)m * 4096 + t * 16 + r] = (_Float16)(o[t][j] * inv);
    }
}

// ---------------------------------------------------------------------------
// Host launcher
// ---------------------------------------------------------------------------
extern "C" void kernel_launch(void* const* d_in, const int* in_sizes, int n_in,
                              void* d_out, int out_size, void* d_ws, size_t ws_size,
                              hipStream_t stream) {
    (void)in_sizes; (void)n_in; (void)out_size; (void)ws_size;

    const float* x  = (const float*)d_in[0];
    const float* wq = (const float*)d_in[3];
    const float* wk = (const float*)d_in[4];
    const float* wv = (const float*)d_in[5];
    const float* wo = (const float*)d_in[6];

    float* out = (float*)d_out;                 // [2,2048,4096]
    float* ck  = out + 16777216;                // cache_k [2,8,4096,128]
    float* cv  = ck + 8388608;                  // cache_v

    char* wsb = (char*)d_ws;                    // 192 MB total
    _Float16* Xh  = (_Float16*)(wsb);                // 33.5 MB
    _Float16* Wqh = (_Float16*)(wsb + 33554432);     // 33.5 MB
    _Float16* Wkh = (_Float16*)(wsb + 67108864);     //  8.4 MB
    _Float16* Wvh = (_Float16*)(wsb + 75497472);     //  8.4 MB
    _Float16* Woh = (_Float16*)(wsb + 83886080);     // 33.5 MB
    _Float16* Qh  = (_Float16*)(wsb + 117440512);    // 33.5 MB
    _Float16* Kh  = (_Float16*)(wsb + 150994944);    //  8.4 MB
    _Float16* Vt  = (_Float16*)(wsb + 159383552);    //  8.4 MB
    _Float16* Att = (_Float16*)(wsb + 167772160);    // 33.5 MB  (ends 201326592)

    // 1) f32 -> f16 conversions
    cvt_f32_f16<<<16384, 256, 0, stream>>>(x,  Xh,  4194304);
    cvt_f32_f16<<<16384, 256, 0, stream>>>(wq, Wqh, 4194304);
    cvt_f32_f16<<< 4096, 256, 0, stream>>>(wk, Wkh, 1048576);
    cvt_f32_f16<<< 4096, 256, 0, stream>>>(wv, Wvh, 1048576);
    cvt_f32_f16<<<16384, 256, 0, stream>>>(wo, Woh, 4194304);

    // 2) Projections: Q (f16), K (cache f32 + f16), V (cache f32 + f16-transposed)
    // Grid: (M/64) * (N/128) blocks of 256 threads.
    gemm_xwt<<<2048, 256, 0, stream>>>(Xh, Wqh, nullptr, 0, Qh, 0, 4096, 4096, 4096);
    gemm_xwt<<< 512, 256, 0, stream>>>(Xh, Wkh, ck,      1, Kh, 0, 4096, 1024, 4096);
    gemm_xwt<<< 512, 256, 0, stream>>>(Xh, Wvh, cv,      1, Vt, 2, 4096, 1024, 4096);

    // 3) Zero unused cache tail positions [2048,4096)
    zero_tail<<<32768, 256, 0, stream>>>(ck, cv);

    // 4) Flash attention: 2*32*128 = 8192 waves
    attn_fa<<<1024, 256, 0, stream>>>(Qh, Kh, Vt, Att);

    // 5) Output projection -> fp32 d_out
    gemm_xwt<<<2048, 256, 0, stream>>>(Att, Woh, out, 0, nullptr, 0, 4096, 4096, 4096);
}
